// Custom_Model_WOQ_Linear_Add_Sequential_24215025615266
// MI455X (gfx1250) — matrix operand order repaired and checked by reading, not compiled, not run on hardware
//
#include <hip/hip_runtime.h>
#include <hip/hip_bf16.h>
#include <stdint.h>

// ---------------------------------------------------------------------------
// Problem constants (from the reference)
// ---------------------------------------------------------------------------
constexpr int Bz   = 4;
constexpr int Sdim = 1024;
constexpr int Kdim = 4096;
constexpr int Ndim = 4096;
constexpr int Mdim = Bz * Sdim;          // 4096 activation rows
constexpr int GroupSz = 128;             // quant group along K
constexpr int NPk  = Ndim / 8;           // packed int4 columns (512)

// GEMM tiling
constexpr int BM = 128;
constexpr int BN = 128;
constexpr int BK = 64;
constexpr int LDT = BK + 8;              // padded LDS row (halves): 144 B, 16B aligned,
                                         // 36-dword row stride -> conflict-free 16-row reads

typedef _Float16 v16h __attribute__((ext_vector_type(16)));
typedef _Float16 v8h  __attribute__((ext_vector_type(8)));
typedef float    v8f  __attribute__((ext_vector_type(8)));

// ---------------------------------------------------------------------------
// CDNA5 async global->LDS copy (tracked by ASYNCcnt, bypasses VGPRs).
// Per ISA 08_async_tensor 4.4: INST_OFFSET is added to BOTH the LDS and the
// global address, so one (lds,global) base pair covers a contiguous 64 B
// row-segment with four b128 copies. The LDS byte address is the low 32 bits
// of the generic pointer (flat LDS aperture keeps the DS offset in [31:0]).
// ---------------------------------------------------------------------------
#define ASYNC_LD16(ldsaddr, gaddr, OFFSTR)                                   \
    asm volatile("global_load_async_to_lds_b128 %0, %1, off offset:" OFFSTR  \
                 :: "v"(ldsaddr), "v"(gaddr) : "memory")

__device__ __forceinline__ void async_copy_row64(uint32_t ldsaddr, const _Float16* gptr)
{
    const uint64_t ga = (uint64_t)(uintptr_t)gptr;
    ASYNC_LD16(ldsaddr, ga, "0");
    ASYNC_LD16(ldsaddr, ga, "16");
    ASYNC_LD16(ldsaddr, ga, "32");
    ASYNC_LD16(ldsaddr, ga, "48");
}

// ---------------------------------------------------------------------------
// Kernel 1: dequantize int4 grouped weights -> fp16 W^T [N, K] via LDS
// transpose. Reads of qweight are coalesced int32s; writes of W^T are
// coalesced b128 rows along K. One 64(K) x 64(N) patch per workgroup; an
// aligned 64-wide K range always lies inside a single 128-wide quant group.
// ---------------------------------------------------------------------------
constexpr int DQ_KT = 64;
constexpr int DQ_NT = 64;
constexpr int DQ_PITCH = 72;             // halves; 144 B rows, 16B aligned

__global__ __launch_bounds__(256)
void dequant_int4_kernel(const int* __restrict__ qweight,
                         const float* __restrict__ scales,
                         const int* __restrict__ qzeros,
                         _Float16* __restrict__ Wt)
{
    __shared__ __attribute__((aligned(16))) _Float16 T[DQ_NT * DQ_PITCH];

    const int k0  = blockIdx.x * DQ_KT;
    const int n0  = blockIdx.y * DQ_NT;
    const int np0 = n0 >> 3;
    const int g   = k0 >> 7;             // constant over the whole patch
    const int t   = threadIdx.x;
    const int c   = t & 7;               // np within patch (0..7)
    const int r0  = t >> 3;              // k row (0..31); also handles r0+32

    const int zp = qzeros[g * NPk + np0 + c];
    float sc[8];
#pragma unroll
    for (int j = 0; j < 8; ++j) sc[j] = scales[g * Ndim + n0 + c * 8 + j];

#pragma unroll
    for (int rr = 0; rr < 2; ++rr) {
        const int r = r0 + rr * 32;
        const int q = qweight[(size_t)(k0 + r) * NPk + np0 + c];
#pragma unroll
        for (int j = 0; j < 8; ++j) {
            const int qv = (q >> (4 * j)) & 0xF;
            const int zv = (zp >> (4 * j)) & 0xF;
            T[(c * 8 + j) * DQ_PITCH + r] = (_Float16)((float)(qv - zv) * sc[j]);
        }
    }
    __syncthreads();

    // write phase: 4 threads per n-row, 32 B (2 x b128) each, coalesced along K
    const int n   = t >> 2;
    const int sgo = (t & 3) * 16;
    const uint4* src = (const uint4*)&T[n * DQ_PITCH + sgo];
    uint4* dst = (uint4*)(Wt + (size_t)(n0 + n) * Kdim + k0 + sgo);
    dst[0] = src[0];
    dst[1] = src[1];
}

// ---------------------------------------------------------------------------
// Kernel 2: fp32 -> fp16 activation conversion (4 elems/thread)
// ---------------------------------------------------------------------------
__global__ __launch_bounds__(256)
void cvt_f32_f16_kernel(const float4* __restrict__ x, _Float16* __restrict__ y)
{
    const size_t i = (size_t)blockIdx.x * blockDim.x + threadIdx.x;
    const float4 v = x[i];
    const size_t o = i * 4;
    y[o + 0] = (_Float16)v.x;
    y[o + 1] = (_Float16)v.y;
    y[o + 2] = (_Float16)v.z;
    y[o + 3] = (_Float16)v.w;
}

// ---------------------------------------------------------------------------
// Kernel 3: WMMA GEMM  C = A[M,K] * (W^T[N,K])^T  with fused epilogues.
//   MODE 0:  add_res = C + bias + add1 + add2   (fp32 + fp16 copies)
//   MODE 1:  out     = add_res * (C + bias + add1)
//
// 8 waves / 128x128 tile / wave owns 64x32 (4x2 accum of 16x16).
// Double-buffered LDS fed by global_load_async_to_lds_b128: tile k+1 streams
// in while WMMAs consume tile k. Each thread issues 8 async b128 copies per
// tile; ASYNCcnt completes in order, so s_wait_asynccnt 8 after issuing the
// next tile guarantees the current tile has landed.
// ---------------------------------------------------------------------------
template <int MODE>
__global__ __launch_bounds__(256)
void gemm_wmma_kernel(const _Float16* __restrict__ A,     // [M,K] fp16
                      const _Float16* __restrict__ Bt,    // [N,K] fp16 (W^T)
                      const float* __restrict__ bias,     // [N]
                      const float* __restrict__ add1,     // [M,N]
                      const float* __restrict__ add2,     // [M,N]  (MODE 0)
                      float* __restrict__ resF,           // MODE0 out / MODE1 in
                      _Float16* __restrict__ resH,        // MODE0 out
                      float* __restrict__ out)            // MODE1 out
{
    __shared__ __attribute__((aligned(16))) _Float16 As[2][BM * LDT];
    __shared__ __attribute__((aligned(16))) _Float16 Bs[2][BN * LDT];

    const int tid  = threadIdx.x;
    const int lane = tid & 31;
    const int wv   = tid >> 5;
    const int lr   = lane & 15;
    const int rh   = lane >> 4;
    const int wm   = (wv >> 2) * 64;
    const int wn   = (wv & 3) * 32;

    const int rowBase = blockIdx.y * BM;
    const int colBase = blockIdx.x * BN;

    // cooperative tile copy: 2 threads per row, 64 B each
    const int ldr = tid >> 1;
    const int seg = (tid & 1) * 32;

    const _Float16* gA = A  + (size_t)(rowBase + ldr) * Kdim + seg;
    const _Float16* gB = Bt + (size_t)(colBase + ldr) * Kdim + seg;

    const uint32_t laA[2] = { (uint32_t)(uintptr_t)&As[0][ldr * LDT + seg],
                              (uint32_t)(uintptr_t)&As[1][ldr * LDT + seg] };
    const uint32_t laB[2] = { (uint32_t)(uintptr_t)&Bs[0][ldr * LDT + seg],
                              (uint32_t)(uintptr_t)&Bs[1][ldr * LDT + seg] };

    v8f acc[4][2] = {};

    // prologue: stream tile 0 into buffer 0
    async_copy_row64(laA[0], gA);
    async_copy_row64(laB[0], gB);

    for (int k0 = 0; k0 < Kdim; k0 += BK) {
        const int cur = (k0 >> 6) & 1;

        if (k0 + BK < Kdim) {
            async_copy_row64(laA[cur ^ 1], gA + k0 + BK);
            async_copy_row64(laB[cur ^ 1], gB + k0 + BK);
            asm volatile("s_wait_asynccnt 0x8" ::: "memory");  // current tile done
        } else {
            asm volatile("s_wait_asynccnt 0x0" ::: "memory");
        }
        __syncthreads();

#pragma unroll
        for (int ks = 0; ks < BK; ks += 32) {
            v16h af[4];
            v16h bf[2];
#pragma unroll
            for (int mi = 0; mi < 4; ++mi) {
                const _Float16* p = &As[cur][(wm + mi * 16 + lr) * LDT + ks + rh * 8];
                v8h lo = *(const v8h*)(p);
                v8h hi = *(const v8h*)(p + 16);
                af[mi] = __builtin_shufflevector(lo, hi, 0,1,2,3,4,5,6,7,
                                                         8,9,10,11,12,13,14,15);
            }
#pragma unroll
            for (int ni = 0; ni < 2; ++ni) {
                const _Float16* p = &Bs[cur][(wn + ni * 16 + lr) * LDT + ks + rh * 8];
                v8h lo = *(const v8h*)(p);
                v8h hi = *(const v8h*)(p + 16);
                bf[ni] = __builtin_shufflevector(lo, hi, 0,1,2,3,4,5,6,7,
                                                         8,9,10,11,12,13,14,15);
            }
#pragma unroll
            for (int mi = 0; mi < 4; ++mi)
#pragma unroll
                for (int ni = 0; ni < 2; ++ni)
                    acc[mi][ni] = __builtin_amdgcn_wmma_f32_16x16x32_f16(
                        false, af[mi], false, bf[ni],
                        (short)0, acc[mi][ni], false, false);
        }
        __syncthreads();   // protect buf[cur] from next iteration's async writes
    }

    // fused epilogue (C layout: VGPR i -> M = i + 8*rh, N = lane&15)
#pragma unroll
    for (int mi = 0; mi < 4; ++mi) {
#pragma unroll
        for (int ni = 0; ni < 2; ++ni) {
            const int col = colBase + wn + ni * 16 + lr;
            const float b = bias[col];
#pragma unroll
            for (int i = 0; i < 8; ++i) {
                const int row = rowBase + wm + mi * 16 + rh * 8 + i;
                const size_t idx = (size_t)row * Ndim + col;
                float v = acc[mi][ni][i] + b;
                if (MODE == 0) {
                    v += add1[idx] + add2[idx];
                    resF[idx] = v;
                    resH[idx] = (_Float16)v;
                } else {
                    out[idx] = resF[idx] * (v + add1[idx]);
                }
            }
        }
    }
}

// ---------------------------------------------------------------------------
// Launch: dequant -> convert -> GEMM1(+adds) -> GEMM2(+final elementwise)
// Workspace layout (needs 160 MB):
//   [0,32MB)    W^T fp16 [N,K]
//   [32,64MB)   X  fp16 [M,K]
//   [64,128MB)  add_res fp32 [M,N]
//   [128,160MB) add_res fp16 [M,N]
// ---------------------------------------------------------------------------
extern "C" void kernel_launch(void* const* d_in, const int* in_sizes, int n_in,
                              void* d_out, int out_size, void* d_ws, size_t ws_size,
                              hipStream_t stream)
{
    const float* inp     = (const float*)d_in[0];
    const int*   qweight = (const int*)  d_in[1];
    const float* scales  = (const float*)d_in[2];
    const int*   qzeros  = (const int*)  d_in[3];
    const float* bias    = (const float*)d_in[4];
    const float* add1    = (const float*)d_in[5];
    const float* add2    = (const float*)d_in[6];
    float*       out     = (float*)d_out;

    char* ws = (char*)d_ws;
    _Float16* Wt    = (_Float16*)(ws);
    _Float16* Xh    = (_Float16*)(ws + (32ull << 20));
    float*    AresF = (float*)   (ws + (64ull << 20));
    _Float16* AresH = (_Float16*)(ws + (128ull << 20));

    dim3 dq_grid(Kdim / DQ_KT, Ndim / DQ_NT);
    dequant_int4_kernel<<<dq_grid, 256, 0, stream>>>(qweight, scales, qzeros, Wt);

    cvt_f32_f16_kernel<<<((size_t)Mdim * Kdim / 4) / 256, 256, 0, stream>>>(
        (const float4*)inp, Xh);

    dim3 grid(Ndim / BN, Mdim / BM);

    gemm_wmma_kernel<0><<<grid, 256, 0, stream>>>(Xh, Wt, bias, add1, add2,
                                                  AresF, AresH, nullptr);

    gemm_wmma_kernel<1><<<grid, 256, 0, stream>>>(AresH, Wt, bias, add1, nullptr,
                                                  AresF, nullptr, out);
}
